// MambaHead_77412490543274
// MI455X (gfx1250) — matrix operand order repaired and checked by reading, not compiled, not run on hardware
//
#include <hip/hip_runtime.h>
#include <hip/hip_bf16.h>
#include <math.h>

// ---------------- problem constants ----------------
#define NLAYERS 3
#define NHEADS  8
#define HEADDIM 64
#define DSTATE  64
#define DCONV   4
#define DINNER  512            // NHEADS*HEADDIM
#define BSZ     8
#define SEQ     2048
#define DFEAT   128
#define NFEAT   15
#define EDIM    256
#define DIP     (2*DINNER + 2*DSTATE + NHEADS)   // 1160
#define DIP_PAD 1184                              // pad to /32 for 64x32 wave tiles
#define CDIM    (DINNER + 2*DSTATE)              // 640
#define M_ROWS  (BSZ*SEQ)                        // 16384

typedef __attribute__((ext_vector_type(16))) __bf16 bf16x16;
typedef __attribute__((ext_vector_type(8)))  __bf16 bf16x8;
typedef __attribute__((ext_vector_type(8)))  float  floatx8;

// ---------------- block reduction helper ----------------
template<int TPB>
__device__ inline float block_sum(float v, float* sdata, int tid) {
    sdata[tid] = v;
    __syncthreads();
#pragma unroll
    for (int s = TPB / 2; s > 0; s >>= 1) {
        if (tid < s) sdata[tid] += sdata[tid + s];
        __syncthreads();
    }
    float r = sdata[0];
    __syncthreads();
    return r;
}

__device__ inline float silu_f(float x) { return x / (1.0f + __expf(-x)); }

// ---------------- 0) feature average (softmax weights) + layernorm -> bf16 A ----------------
__global__ __launch_bounds__(DFEAT)
void avg_ln_kernel(const float* __restrict__ feature, const float* __restrict__ w_avg,
                   const float* __restrict__ g, const float* __restrict__ b,
                   __bf16* __restrict__ A0) {
    const int row = blockIdx.x;           // b*SEQ + s
    const int d   = threadIdx.x;          // 0..127
    float w[NFEAT];
    float mx = -1e30f;
#pragma unroll
    for (int f = 0; f < NFEAT; f++) { w[f] = w_avg[f]; mx = fmaxf(mx, w[f]); }
    float sum = 0.f;
#pragma unroll
    for (int f = 0; f < NFEAT; f++) { w[f] = __expf(w[f] - mx); sum += w[f]; }
    const float inv = 1.0f / sum;
    float x = 0.f;
#pragma unroll
    for (int f = 0; f < NFEAT; f++)
        x += feature[((size_t)f * M_ROWS + row) * DFEAT + d] * (w[f] * inv);

    __shared__ float sdata[DFEAT];
    const float mean = block_sum<DFEAT>(x, sdata, d) * (1.0f / DFEAT);
    const float dv   = x - mean;
    const float var  = block_sum<DFEAT>(dv * dv, sdata, d) * (1.0f / DFEAT);
    const float o    = dv * rsqrtf(var + 1e-5f) * g[d] + b[d];
    A0[(size_t)row * DFEAT + d] = (__bf16)o;
}

// ---------------- 1) weight prep: fp32 W[K][N] -> bf16 Wt[Npad][K] (transposed, padded) ----------------
__global__ __launch_bounds__(256)
void prep_weight_kernel(const float* __restrict__ W, __bf16* __restrict__ Wt,
                        int K, int N, int Npad) {
    int idx = blockIdx.x * 256 + threadIdx.x;
    int total = Npad * K;
    if (idx >= total) return;
    int n = idx / K, k = idx - n * K;
    float v = (n < N) ? W[(size_t)k * N + n] : 0.0f;
    Wt[idx] = (__bf16)v;
}

// ---------------- 2) WMMA GEMM: 64x32 output tile per wave ----------------
// C[M][Npad] (+= / = bias+) A[M][K](bf16) * Wt[Npad][K](bf16, N-major)
// fragment layouts per CDNA5 ISA 7.12.2; 8 independent v_wmma per K-step.
__global__ __launch_bounds__(32)
void wmma_gemm_kernel(const __bf16* __restrict__ A, const __bf16* __restrict__ Bt,
                      float* __restrict__ C, const float* __restrict__ bias,
                      int accumulate, int M, int K, int Npad) {
    const int n0   = blockIdx.x * 32;
    const int m0   = blockIdx.y * 64;
    const int lane = threadIdx.x;          // 0..31
    const int half = lane >> 4;            // 0 or 1
    const int l16  = lane & 15;

    const __bf16* Ap0 = A + (size_t)(m0 + 0 * 16 + l16) * K;
    const __bf16* Ap1 = A + (size_t)(m0 + 1 * 16 + l16) * K;
    const __bf16* Ap2 = A + (size_t)(m0 + 2 * 16 + l16) * K;
    const __bf16* Ap3 = A + (size_t)(m0 + 3 * 16 + l16) * K;
    const __bf16* Bp0 = Bt + (size_t)(n0 + 0 * 16 + l16) * K;
    const __bf16* Bp1 = Bt + (size_t)(n0 + 1 * 16 + l16) * K;

    floatx8 acc[4][2];
#pragma unroll
    for (int i = 0; i < 4; i++)
#pragma unroll
        for (int j = 0; j < 2; j++) acc[i][j] = (floatx8){};

    const int aoff_lo = half * 8;          // A elems 0..7 : K[k0+half*8 ..]
    const int aoff_hi = 16 + half * 8;     // A elems 8..15: K[k0+16+half*8 ..]
    const int boff    = half * 8;          // B elems 0..7: K[k0+half*8 ..], 8..15: K[k0+16+half*8 ..]

    for (int k0 = 0; k0 < K; k0 += 32) {
        __builtin_prefetch((const void*)(Ap0 + k0 + 256), 0, 0);
        __builtin_prefetch((const void*)(Bp0 + k0 + 256), 0, 0);
        bf16x16 a[4], b[2];
        {
            bf16x8 lo, hi;
            lo = *(const bf16x8*)(Ap0 + k0 + aoff_lo); hi = *(const bf16x8*)(Ap0 + k0 + aoff_hi);
#pragma unroll
            for (int e = 0; e < 8; e++) { a[0][e] = lo[e]; a[0][e + 8] = hi[e]; }
            lo = *(const bf16x8*)(Ap1 + k0 + aoff_lo); hi = *(const bf16x8*)(Ap1 + k0 + aoff_hi);
#pragma unroll
            for (int e = 0; e < 8; e++) { a[1][e] = lo[e]; a[1][e + 8] = hi[e]; }
            lo = *(const bf16x8*)(Ap2 + k0 + aoff_lo); hi = *(const bf16x8*)(Ap2 + k0 + aoff_hi);
#pragma unroll
            for (int e = 0; e < 8; e++) { a[2][e] = lo[e]; a[2][e + 8] = hi[e]; }
            lo = *(const bf16x8*)(Ap3 + k0 + aoff_lo); hi = *(const bf16x8*)(Ap3 + k0 + aoff_hi);
#pragma unroll
            for (int e = 0; e < 8; e++) { a[3][e] = lo[e]; a[3][e + 8] = hi[e]; }
            lo = *(const bf16x8*)(Bp0 + k0 + boff);    hi = *(const bf16x8*)(Bp0 + k0 + boff + 16);
#pragma unroll
            for (int e = 0; e < 8; e++) { b[0][e] = lo[e]; b[0][e + 8] = hi[e]; }
            lo = *(const bf16x8*)(Bp1 + k0 + boff);    hi = *(const bf16x8*)(Bp1 + k0 + boff + 16);
#pragma unroll
            for (int e = 0; e < 8; e++) { b[1][e] = lo[e]; b[1][e + 8] = hi[e]; }
        }
#pragma unroll
        for (int i = 0; i < 4; i++)
#pragma unroll
            for (int j = 0; j < 2; j++)
                acc[i][j] = __builtin_amdgcn_wmma_f32_16x16x32_bf16(
                    false, a[i], false, b[j], (short)0, acc[i][j], false, false);
    }

    // D 16x16 f32 per tile: lane(half,n) vgpr r -> row half*8+r, col n
#pragma unroll
    for (int i = 0; i < 4; i++) {
#pragma unroll
        for (int j = 0; j < 2; j++) {
            const int col = n0 + j * 16 + l16;
            const float bv = bias ? bias[col] : 0.0f;
            float* Crow = C + (size_t)(m0 + i * 16 + half * 8) * Npad + col;
#pragma unroll
            for (int r = 0; r < 8; r++) {
                float v = acc[i][j][r] + bv;
                if (accumulate) v += Crow[(size_t)r * Npad];
                Crow[(size_t)r * Npad] = v;
            }
        }
    }
}

// ---------------- elementwise epilogue: bias + exact gelu ----------------
__global__ __launch_bounds__(256)
void bias_gelu_kernel(const float* __restrict__ t, const float* __restrict__ bias,
                      float* __restrict__ o, int N) {
    size_t i = (size_t)blockIdx.x * 256 + threadIdx.x;
    float v = t[i] + bias[i % N];
    o[i] = 0.5f * v * (1.0f + erff(v * 0.70710678118654752f));
}

// ---------------- per-layer: rmsnorm -> bf16 ----------------
__global__ __launch_bounds__(EDIM)
void rms_bf16_kernel(const float* __restrict__ x, const float* __restrict__ w,
                     __bf16* __restrict__ o) {
    const size_t row = blockIdx.x;
    const int d = threadIdx.x;
    float v = x[row * EDIM + d];
    __shared__ float sdata[EDIM];
    float ssq = block_sum<EDIM>(v * v, sdata, d);
    float r = rsqrtf(ssq * (1.0f / EDIM) + 1e-5f);
    o[row * EDIM + d] = (__bf16)(v * r * w[d]);
}

// ---------------- depthwise conv (DCONV=4) + silu on xBC, softplus dt, dA ----------------
__global__ __launch_bounds__(256)
void conv_dt_kernel(const float* __restrict__ zx,      // [M][DIP_PAD]
                    const float* __restrict__ convW,   // [CDIM][DCONV]
                    const float* __restrict__ convB,   // [CDIM]
                    const float* __restrict__ dtb,     // [NHEADS]
                    const float* __restrict__ Alog,    // [NHEADS]
                    float* __restrict__ xconv,         // [M][CDIM]
                    float* __restrict__ dtA)           // [M][16]  (dt | dA)
{
    const size_t row = blockIdx.x;
    const int s = (int)(row % SEQ);
    for (int c = threadIdx.x; c < CDIM; c += 256) {
        float acc = convB[c];
#pragma unroll
        for (int k = 0; k < DCONV; k++) {
            int sp = s + k - (DCONV - 1);
            if (sp >= 0)
                acc += convW[c * DCONV + k] *
                       zx[(row - (size_t)(DCONV - 1 - k)) * DIP_PAD + DINNER + c];
        }
        xconv[row * CDIM + c] = silu_f(acc);
    }
    if (threadIdx.x < NHEADS) {
        int h = threadIdx.x;
        float raw = zx[row * DIP_PAD + 2 * DINNER + 2 * DSTATE + h] + dtb[h];
        float dt = (raw > 20.0f) ? raw : log1pf(__expf(raw));
        float A = -__expf(Alog[h]);
        dtA[row * 16 + h]     = dt;
        dtA[row * 16 + 8 + h] = __expf(dt * A);
    }
}

// ---------------- sequential selective scan ----------------
// one block of 256 threads per (batch, head); DSTATE split 4-way across
// thread groups (h[16] per thread) to shorten the serial VALU chain;
// LDS partial-sum reduction for y.
__global__ __launch_bounds__(256)
void scan_kernel(const float* __restrict__ xconv,  // [M][CDIM]
                 const float* __restrict__ dtA,    // [M][16]
                 const float* __restrict__ Dp,     // [NHEADS]
                 float* __restrict__ y)            // [M][DINNER]
{
    const int bh = blockIdx.x;
    const int b  = bh / NHEADS;
    const int hd = bh % NHEADS;
    const int tid = threadIdx.x;
    const int p   = tid & 63;              // head-dim index
    const int q   = tid >> 6;              // state-group 0..3
    const int nbase = q * 16;

    float h[16];
#pragma unroll
    for (int n = 0; n < 16; n++) h[n] = 0.f;

    __shared__ float Bs[DSTATE];
    __shared__ float Cs[DSTATE];
    __shared__ float red[256];
    const float Dcoef = Dp[hd];

    for (int t = 0; t < SEQ; t++) {
        const size_t row = (size_t)b * SEQ + t;
        const float* xc = xconv + row * CDIM;
        if (tid < DSTATE)            Bs[tid]      = xc[DINNER + tid];
        else if (tid < 2 * DSTATE)   Cs[tid - 64] = xc[DINNER + DSTATE + (tid - 64)];
        __syncthreads();
        const float dt = dtA[row * 16 + hd];
        const float dA = dtA[row * 16 + 8 + hd];
        const float xt = xc[hd * HEADDIM + p];
        const float coef = dt * xt;
        float acc = 0.f;
#pragma unroll
        for (int n = 0; n < 16; n++) {
            h[n] = h[n] * dA + coef * Bs[nbase + n];
            acc += h[n] * Cs[nbase + n];
        }
        red[tid] = acc;
        __syncthreads();
        if (q == 0)
            y[row * DINNER + hd * HEADDIM + p] =
                red[p] + red[64 + p] + red[128 + p] + red[192 + p] + Dcoef * xt;
    }
}

// ---------------- gate (y * silu(z)) + group rmsnorm -> bf16 ----------------
__global__ __launch_bounds__(DINNER)
void gate_rms_kernel(const float* __restrict__ y, const float* __restrict__ zx,
                     const float* __restrict__ gnw, __bf16* __restrict__ o) {
    const size_t row = blockIdx.x;
    const int d = threadIdx.x;
    float z = zx[row * DIP_PAD + d];
    float g = y[row * DINNER + d] * silu_f(z);
    __shared__ float sdata[DINNER];
    float ssq = block_sum<DINNER>(g * g, sdata, d);
    float r = rsqrtf(ssq * (1.0f / DINNER) + 1e-5f);
    o[row * DINNER + d] = (__bf16)(g * r * gnw[d]);
}

// ---------------- final: rmsnorm then layernorm -> bf16 ----------------
__global__ __launch_bounds__(EDIM)
void final_norm_kernel(const float* __restrict__ x, const float* __restrict__ nw,
                       const float* __restrict__ g, const float* __restrict__ b,
                       __bf16* __restrict__ o) {
    const size_t row = blockIdx.x;
    const int d = threadIdx.x;
    float v = x[row * EDIM + d];
    __shared__ float sdata[EDIM];
    float ssq = block_sum<EDIM>(v * v, sdata, d);
    float t = v * rsqrtf(ssq * (1.0f / EDIM) + 1e-5f) * nw[d];
    float mean = block_sum<EDIM>(t, sdata, d) * (1.0f / EDIM);
    float dv = t - mean;
    float var = block_sum<EDIM>(dv * dv, sdata, d) * (1.0f / EDIM);
    o[row * EDIM + d] = (__bf16)(dv * rsqrtf(var + 1e-5f) * g[d] + b[d]);
}

// ---------------- final projection 256 -> 5 ----------------
__global__ __launch_bounds__(EDIM)
void final_out_kernel(const float* __restrict__ hg, const float* __restrict__ W2,
                      const float* __restrict__ b2, float* __restrict__ out) {
    const size_t row = blockIdx.x;
    const int d = threadIdx.x;
    float hv = hg[row * EDIM + d];
    __shared__ float sdata[EDIM];
#pragma unroll
    for (int j = 0; j < 5; j++) {
        float s = block_sum<EDIM>(hv * W2[d * 5 + j], sdata, d);
        if (d == 0) out[row * 5 + j] = s + b2[j];
    }
}

// =====================================================================
extern "C" void kernel_launch(void* const* d_in, const int* in_sizes, int n_in,
                              void* d_out, int out_size, void* d_ws, size_t ws_size,
                              hipStream_t stream) {
    const float* feature    = (const float*)d_in[0];
    const float* w_avg      = (const float*)d_in[1];
    const float* inproj_ln_g= (const float*)d_in[2];
    const float* inproj_ln_b= (const float*)d_in[3];
    const float* inproj_W   = (const float*)d_in[4];
    const float* inproj_b   = (const float*)d_in[5];
    const float* rms_w      = (const float*)d_in[6];
    const float* m_inW      = (const float*)d_in[7];
    const float* m_convW    = (const float*)d_in[8];
    const float* m_convB    = (const float*)d_in[9];
    const float* m_dtb      = (const float*)d_in[10];
    const float* m_Alog     = (const float*)d_in[11];
    const float* m_D        = (const float*)d_in[12];
    const float* m_gnw      = (const float*)d_in[13];
    const float* m_outW     = (const float*)d_in[14];
    const float* norm_w     = (const float*)d_in[15];
    const float* mlp_ln_g   = (const float*)d_in[16];
    const float* mlp_ln_b   = (const float*)d_in[17];
    const float* mlp_W1     = (const float*)d_in[18];
    const float* mlp_b1     = (const float*)d_in[19];
    const float* mlp_W2     = (const float*)d_in[20];
    const float* mlp_b2     = (const float*)d_in[21];
    float* out = (float*)d_out;

    // ---- carve workspace ----
    size_t off = 0;
    char* base = (char*)d_ws;
    auto carve = [&](size_t bytes) -> void* {
        void* p = base + off;
        off += (bytes + 255) & ~(size_t)255;
        return p;
    };
    float*  x     = (float*) carve((size_t)M_ROWS * EDIM * 4);      // residual stream
    __bf16* Abf   = (__bf16*)carve((size_t)M_ROWS * DINNER * 2);    // bf16 activations (max 512 wide)
    float*  gtmp  = (float*) carve((size_t)M_ROWS * DIP_PAD * 4);   // gemm outputs / zxbcdt
    float*  xconv = (float*) carve((size_t)M_ROWS * CDIM * 4);
    float*  dtA   = (float*) carve((size_t)M_ROWS * 16 * 4);
    float*  ybuf  = (float*) carve((size_t)M_ROWS * DINNER * 4);    // scan out; reused for gelu out
    __bf16* Wt    = (__bf16*)carve((size_t)DIP_PAD * DINNER * 2);   // transposed bf16 weights

    const int EW_BLOCKS = (M_ROWS * EDIM) / 256;

    // 0) feature average + layernorm -> bf16  (A: M x 128)
    avg_ln_kernel<<<M_ROWS, DFEAT, 0, stream>>>(feature, w_avg, inproj_ln_g, inproj_ln_b, Abf);

    // 1) inproj GEMM (K=128, N=256), bias fused -> x
    prep_weight_kernel<<<(EDIM * DFEAT + 255) / 256, 256, 0, stream>>>(inproj_W, Wt, DFEAT, EDIM, EDIM);
    wmma_gemm_kernel<<<dim3(EDIM / 32, M_ROWS / 64), 32, 0, stream>>>(
        Abf, Wt, x, inproj_b, /*accumulate=*/0, M_ROWS, DFEAT, EDIM);

    // 2) mamba2 layers
    for (int i = 0; i < NLAYERS; i++) {
        const float* rms_w_i  = rms_w  + (size_t)i * EDIM;
        const float* inW_i    = m_inW  + (size_t)i * EDIM * DIP;
        const float* convW_i  = m_convW+ (size_t)i * CDIM * DCONV;
        const float* convB_i  = m_convB+ (size_t)i * CDIM;
        const float* dtb_i    = m_dtb  + (size_t)i * NHEADS;
        const float* Alog_i   = m_Alog + (size_t)i * NHEADS;
        const float* D_i      = m_D    + (size_t)i * NHEADS;
        const float* gnw_i    = m_gnw  + (size_t)i * DINNER;
        const float* outW_i   = m_outW + (size_t)i * DINNER * EDIM;

        // rmsnorm(x) -> bf16
        rms_bf16_kernel<<<M_ROWS, EDIM, 0, stream>>>(x, rms_w_i, Abf);
        // zxbcdt = xn @ inW  (K=256, N=1160 padded to 1184)
        prep_weight_kernel<<<(DIP_PAD * EDIM + 255) / 256, 256, 0, stream>>>(inW_i, Wt, EDIM, DIP, DIP_PAD);
        wmma_gemm_kernel<<<dim3(DIP_PAD / 32, M_ROWS / 64), 32, 0, stream>>>(
            Abf, Wt, gtmp, nullptr, 0, M_ROWS, EDIM, DIP_PAD);
        // conv + silu + dt/dA
        conv_dt_kernel<<<M_ROWS, 256, 0, stream>>>(gtmp, convW_i, convB_i, dtb_i, Alog_i, xconv, dtA);
        // sequential scan (64 blocks x 256 threads, 4-way state split)
        scan_kernel<<<BSZ * NHEADS, 256, 0, stream>>>(xconv, dtA, D_i, ybuf);
        // gate + group rmsnorm -> bf16 (M x 512)
        gate_rms_kernel<<<M_ROWS, DINNER, 0, stream>>>(ybuf, gtmp, gnw_i, Abf);
        // out projection (K=512, N=256), residual accumulate into x
        prep_weight_kernel<<<(EDIM * DINNER + 255) / 256, 256, 0, stream>>>(outW_i, Wt, DINNER, EDIM, EDIM);
        wmma_gemm_kernel<<<dim3(EDIM / 32, M_ROWS / 64), 32, 0, stream>>>(
            Abf, Wt, x, nullptr, /*accumulate=*/1, M_ROWS, DINNER, EDIM);
    }

    // 3) final rmsnorm + layernorm -> bf16
    final_norm_kernel<<<M_ROWS, EDIM, 0, stream>>>(x, norm_w, mlp_ln_g, mlp_ln_b, Abf);
    // MLP W1 (K=256, N=256) + bias + gelu
    prep_weight_kernel<<<(EDIM * EDIM + 255) / 256, 256, 0, stream>>>(mlp_W1, Wt, EDIM, EDIM, EDIM);
    wmma_gemm_kernel<<<dim3(EDIM / 32, M_ROWS / 64), 32, 0, stream>>>(
        Abf, Wt, gtmp, nullptr, 0, M_ROWS, EDIM, EDIM);
    bias_gelu_kernel<<<EW_BLOCKS, 256, 0, stream>>>(gtmp, mlp_b1, ybuf, EDIM);
    // final 256 -> 5
    final_out_kernel<<<M_ROWS, EDIM, 0, stream>>>(ybuf, mlp_W2, mlp_b2, out);
}
